// adaptiveUPCSCLoss_66056597012480
// MI455X (gfx1250) — 8-bit matrix operand order repaired and checked by reading, compiled, not run on hardware
//
#include <hip/hip_runtime.h>
#include <hip/hip_bf16.h>
#include <math.h>

typedef __attribute__((ext_vector_type(16))) __bf16 v16bf;
typedef __attribute__((ext_vector_type(8)))  float  v8f;
typedef __attribute__((ext_vector_type(8)))  int    v8i;
typedef __bf16 bf16_t;

#define NROWS 8192
#define DDIM  512
#define CCLS  1000
#define CPAD  1024
#define LOWT  (1.0f / 1000.0f)
#define SLICES 32
#define PAIRS_PER_SLICE ((NROWS / 32) / SLICES)

__global__ void k_init(int* counters) {
    if (threadIdx.x < 2) counters[threadIdx.x] = 0;
}

__global__ void k_cast_pad_raw(const float* __restrict__ src, bf16_t* __restrict__ dst) {
    int idx = blockIdx.x * 256 + threadIdx.x;
    int c = idx >> 9, k = idx & 511;
    float v = (c < CCLS) ? src[(size_t)c * DDIM + k] : 0.0f;
    dst[idx] = (bf16_t)v;
}

__global__ void k_cast_w(const float* __restrict__ src, bf16_t* __restrict__ dst) {
    int idx = blockIdx.x * 256 + threadIdx.x;
    dst[idx] = (bf16_t)src[idx];
}

__global__ void k_rownorm_to_bf16(const float* __restrict__ src, bf16_t* __restrict__ dst) {
    int row  = blockIdx.x;
    int lane = threadIdx.x;
    const float* p = src + (size_t)row * DDIM;
    float v[16];
    float ss = 0.f;
#pragma unroll
    for (int j = 0; j < 16; ++j) { v[j] = p[lane + 32 * j]; ss += v[j] * v[j]; }
#pragma unroll
    for (int off = 16; off >= 1; off >>= 1) ss += __shfl_xor(ss, off, 32);
    float inv = 1.0f / fmaxf(sqrtf(ss), 1e-12f);
    bf16_t* q = dst + (size_t)row * DDIM;
#pragma unroll
    for (int j = 0; j < 16; ++j) q[lane + 32 * j] = (bf16_t)(v[j] * inv);
}

__global__ void k_mm_abt_bf16(const bf16_t* __restrict__ A, const bf16_t* __restrict__ B,
                              float* __restrict__ Cm, int ldc) {
    int tm = blockIdx.x, tn = blockIdx.y;
    int lane = threadIdx.x;
    int sub = lane & 15, g = lane >> 4;
    const bf16_t* ap = A + (size_t)(tm * 16 + sub) * DDIM + g * 16;
    const bf16_t* bp = B + (size_t)(tn * 16 + sub) * DDIM + g * 16;
    v8f acc = {};
#pragma unroll
    for (int kt = 0; kt < DDIM / 32; ++kt) {
        v16bf a = *(const v16bf*)(ap + kt * 32);
        v16bf b = *(const v16bf*)(bp + kt * 32);
        acc = __builtin_amdgcn_wmma_f32_16x16x32_bf16(false, a, false, b,
                                                      (short)0, acc, false, false);
    }
#pragma unroll
    for (int r = 0; r < 8; ++r)
        Cm[(size_t)(tm * 16 + g * 8 + r) * ldc + tn * 16 + sub] = acc[r];
}

__global__ void k_class(const float* __restrict__ prob, const int* __restrict__ conf,
                        unsigned char* __restrict__ pc, int* __restrict__ top1,
                        int* __restrict__ counters) {
    int i = blockIdx.x;
    int lane = threadIdx.x;
    const float* p = prob + (size_t)i * CCLS;
    float m = -__builtin_inff();
    int   mi = 0x7fffffff;
    for (int c = lane; c < CCLS; c += 32) {
        float v = p[c];
        if (v > m || (v == m && c < mi)) { m = v; mi = c; }
    }
#pragma unroll
    for (int off = 16; off >= 1; off >>= 1) {
        float om = __shfl_xor(m, off, 32);
        int   oi = __shfl_xor(mi, off, 32);
        if (om > m || (om == m && oi < mi)) { m = om; mi = oi; }
    }
    int cf = conf[i];
    unsigned char* q = pc + (size_t)i * CPAD;
    int cnt = 0;
    for (int c = lane; c < CPAD; c += 32) {
        int val = 0;
        if (c < CCLS) {
            int cand = p[c] > LOWT;
            val  = cf ? (c == mi) : cand;
            cnt += (cand && !cf);
        }
        q[c] = (unsigned char)val;
    }
#pragma unroll
    for (int off = 16; off >= 1; off >>= 1) cnt += __shfl_xor(cnt, off, 32);
    if (lane == 0) {
        top1[i] = mi;
        atomicAdd(&counters[0], cnt);
        atomicAdd(&counters[1], cf ? 0 : 1);
    }
}

__global__ void k_pos(const float* __restrict__ prob, const float* __restrict__ pred,
                      const int* __restrict__ top1, const int* __restrict__ conf,
                      float* __restrict__ pos) {
    int i = blockIdx.x;
    int lane = threadIdx.x;
    const float* p  = prob + (size_t)i * CCLS;
    const float* pr = pred + (size_t)i * CPAD;
    float sc = 0.f;
    for (int c = lane; c < CCLS; c += 32) {
        float w = p[c];
        if (w > LOWT) sc += w * pr[c];
    }
#pragma unroll
    for (int off = 16; off >= 1; off >>= 1) sc += __shfl_xor(sc, off, 32);
    if (lane == 0) pos[i] = conf[i] ? pr[top1[i]] : sc;
}

__global__ void __launch_bounds__(32)
k_sim(const bf16_t* __restrict__ feat, const unsigned char* __restrict__ pc,
      float* __restrict__ partM, float* __restrict__ partS) {
    int tp    = blockIdx.x;
    int slice = blockIdx.y;
    int lane  = threadIdx.x;
    int sub = lane & 15, g = lane >> 4;

    const bf16_t* a0 = feat + (size_t)(tp * 32 +      sub) * DDIM + g * 16;
    const bf16_t* a1 = feat + (size_t)(tp * 32 + 16 + sub) * DDIM + g * 16;
    const unsigned char* apc0 = pc + (size_t)(tp * 32 +      sub) * CPAD;
    const unsigned char* apc1 = pc + (size_t)(tp * 32 + 16 + sub) * CPAD;

    float m[16], s[16];
#pragma unroll
    for (int r = 0; r < 16; ++r) { m[r] = -__builtin_inff(); s[r] = 0.f; }

    for (int jp = slice * PAIRS_PER_SLICE; jp < (slice + 1) * PAIRS_PER_SLICE; ++jp) {
        int j0 = jp * 32;
        const bf16_t* b0 = feat + (size_t)(j0 +      sub) * DDIM + g * 16;
        const bf16_t* b1 = feat + (size_t)(j0 + 16 + sub) * DDIM + g * 16;
        v8f acc00 = {}, acc01 = {}, acc10 = {}, acc11 = {};
#pragma unroll
        for (int kt = 0; kt < DDIM / 32; ++kt) {
            v16bf A0 = *(const v16bf*)(a0 + kt * 32);
            v16bf A1 = *(const v16bf*)(a1 + kt * 32);
            v16bf B0 = *(const v16bf*)(b0 + kt * 32);
            v16bf B1 = *(const v16bf*)(b1 + kt * 32);
            acc00 = __builtin_amdgcn_wmma_f32_16x16x32_bf16(false, A0, false, B0, (short)0, acc00, false, false);
            acc01 = __builtin_amdgcn_wmma_f32_16x16x32_bf16(false, A0, false, B1, (short)0, acc01, false, false);
            acc10 = __builtin_amdgcn_wmma_f32_16x16x32_bf16(false, A1, false, B0, (short)0, acc10, false, false);
            acc11 = __builtin_amdgcn_wmma_f32_16x16x32_bf16(false, A1, false, B1, (short)0, acc11, false, false);
        }
        const unsigned char* bpc0 = pc + (size_t)(j0 +      sub) * CPAD;
        const unsigned char* bpc1 = pc + (size_t)(j0 + 16 + sub) * CPAD;
        v8i o00 = {}, o01 = {}, o10 = {}, o11 = {};
#pragma unroll
        for (int kt = 0; kt < CPAD / 64; ++kt) {
            v8i A80, A81, B80, B81;
#pragma unroll
            for (int c4 = 0; c4 < 2; ++c4) {
                uint4 d0 = *(const uint4*)(apc0 + kt * 64 + 32 * c4 + 16 * g);
                uint4 d1 = *(const uint4*)(apc1 + kt * 64 + 32 * c4 + 16 * g);
                A80[4 * c4] = (int)d0.x;  A80[4 * c4 + 1] = (int)d0.y;
                A80[4 * c4 + 2] = (int)d0.z;  A80[4 * c4 + 3] = (int)d0.w;
                A81[4 * c4] = (int)d1.x;  A81[4 * c4 + 1] = (int)d1.y;
                A81[4 * c4 + 2] = (int)d1.z;  A81[4 * c4 + 3] = (int)d1.w;
            }
#pragma unroll
            for (int c2 = 0; c2 < 2; ++c2) {
                uint4 d0 = *(const uint4*)(bpc0 + kt * 64 + 32 * c2 + 16 * g);
                uint4 d1 = *(const uint4*)(bpc1 + kt * 64 + 32 * c2 + 16 * g);
                B80[4 * c2] = (int)d0.x; B80[4 * c2 + 1] = (int)d0.y;
                B80[4 * c2 + 2] = (int)d0.z; B80[4 * c2 + 3] = (int)d0.w;
                B81[4 * c2] = (int)d1.x; B81[4 * c2 + 1] = (int)d1.y;
                B81[4 * c2 + 2] = (int)d1.z; B81[4 * c2 + 3] = (int)d1.w;
            }
            o00 = __builtin_amdgcn_wmma_i32_16x16x64_iu8(false, A80, false, B80, o00, false, false);
            o01 = __builtin_amdgcn_wmma_i32_16x16x64_iu8(false, A80, false, B81, o01, false, false);
            o10 = __builtin_amdgcn_wmma_i32_16x16x64_iu8(false, A81, false, B80, o10, false, false);
            o11 = __builtin_amdgcn_wmma_i32_16x16x64_iu8(false, A81, false, B81, o11, false, false);
        }
        auto upd = [&](const v8f& a, const v8i& o, int base) {
#pragma unroll
            for (int r = 0; r < 8; ++r) {
                float v = a[r];
                if (o[r] == 0 && v >= 1e-6f) {
                    if (v > m[base + r]) {
                        s[base + r] = s[base + r] * __expf(m[base + r] - v) + 1.0f;
                        m[base + r] = v;
                    } else {
                        s[base + r] += __expf(v - m[base + r]);
                    }
                }
            }
        };
        upd(acc00, o00, 0);
        upd(acc01, o01, 0);
        upd(acc10, o10, 8);
        upd(acc11, o11, 8);
    }
#pragma unroll
    for (int off = 1; off < 16; off <<= 1) {
#pragma unroll
        for (int r = 0; r < 16; ++r) {
            float om = __shfl_xor(m[r], off, 32);
            float os = __shfl_xor(s[r], off, 32);
            if (om != -__builtin_inff()) {
                if (m[r] == -__builtin_inff()) { m[r] = om; s[r] = os; }
                else {
                    float M = fmaxf(m[r], om);
                    s[r] = s[r] * __expf(m[r] - M) + os * __expf(om - M);
                    m[r] = M;
                }
            }
        }
    }
    if (sub == 0) {
#pragma unroll
        for (int rt = 0; rt < 2; ++rt) {
#pragma unroll
            for (int r = 0; r < 8; ++r) {
                int row = tp * 32 + rt * 16 + g * 8 + r;
                partM[(size_t)row * SLICES + slice] = m[rt * 8 + r];
                partS[(size_t)row * SLICES + slice] = s[rt * 8 + r];
            }
        }
    }
}

__global__ void k_rowlse(const float* __restrict__ partM, const float* __restrict__ partS,
                         const float* __restrict__ pos, float* __restrict__ rowloss) {
    int row = blockIdx.x * blockDim.x + threadIdx.x;
    if (row >= NROWS) return;
    float M = -__builtin_inff(), S = 0.f;
    for (int j = 0; j < SLICES; ++j) {
        float om = partM[(size_t)row * SLICES + j];
        float os = partS[(size_t)row * SLICES + j];
        if (om != -__builtin_inff()) {
            if (M == -__builtin_inff()) { M = om; S = os; }
            else {
                float MM = fmaxf(M, om);
                S = S * __expf(M - MM) + os * __expf(om - MM);
                M = MM;
            }
        }
    }
    float p  = pos[row];
    float MM = fmaxf(M, p);
    float S2 = ((M == -__builtin_inff()) ? 0.f : S * __expf(M - MM)) + __expf(p - MM);
    rowloss[row] = (MM + logf(S2)) - p;
}

__global__ void k_final(const float* __restrict__ rowloss, const int* __restrict__ counters,
                        float* __restrict__ out) {
    __shared__ float sm[256];
    int t = threadIdx.x;
    float acc = 0.f;
    for (int i = t; i < NROWS; i += 256) acc += rowloss[i];
    sm[t] = acc;
    __syncthreads();
    for (int off = 128; off >= 1; off >>= 1) {
        if (t < off) sm[t] += sm[t + off];
        __syncthreads();
    }
    if (t == 0) {
        out[0] = sm[0] / (float)NROWS;
        int denom = counters[1];
        out[1] = (denom > 0) ? (float)counters[0] / (float)denom : 0.0f;
    }
}

extern "C" void kernel_launch(void* const* d_in, const int* in_sizes, int n_in,
                              void* d_out, int out_size, void* d_ws, size_t ws_size,
                              hipStream_t stream) {
    const float* feature   = (const float*)d_in[0];
    const float* proxy_raw = (const float*)d_in[1];
    const float* cPw       = (const float*)d_in[2];
    const float* prob      = (const float*)d_in[3];
    const int*   conf      = (const int*)d_in[4];
    float* out = (float*)d_out;

    char* ws = (char*)d_ws;
    size_t off = 0;
    auto alloc = [&](size_t b) {
        size_t o = off;
        off = (off + b + 255) & ~(size_t)255;
        return o;
    };
    bf16_t* featb    = (bf16_t*)(ws + alloc((size_t)NROWS * DDIM * 2));
    bf16_t* rawb     = (bf16_t*)(ws + alloc((size_t)CPAD * DDIM * 2));
    bf16_t* wb       = (bf16_t*)(ws + alloc((size_t)DDIM * DDIM * 2));
    float*  proxytmp = (float*) (ws + alloc((size_t)CPAD * DDIM * 4));
    bf16_t* proxyb   = (bf16_t*)(ws + alloc((size_t)CPAD * DDIM * 2));
    unsigned char* pcb = (unsigned char*)(ws + alloc((size_t)NROWS * CPAD));
    float*  pred     = (float*) (ws + alloc((size_t)NROWS * CPAD * 4));
    int*    top1     = (int*)   (ws + alloc((size_t)NROWS * 4));
    float*  pos      = (float*) (ws + alloc((size_t)NROWS * 4));
    float*  partM    = (float*) (ws + alloc((size_t)NROWS * SLICES * 4));
    float*  partS    = (float*) (ws + alloc((size_t)NROWS * SLICES * 4));
    float*  rowloss  = (float*) (ws + alloc((size_t)NROWS * 4));
    int*    counters = (int*)   (ws + alloc(256));

    k_init<<<1, 32, 0, stream>>>(counters);
    k_cast_pad_raw<<<(CPAD * DDIM) / 256, 256, 0, stream>>>(proxy_raw, rawb);
    k_cast_w<<<(DDIM * DDIM) / 256, 256, 0, stream>>>(cPw, wb);
    k_rownorm_to_bf16<<<NROWS, 32, 0, stream>>>(feature, featb);
    k_mm_abt_bf16<<<dim3(CPAD / 16, DDIM / 16), 32, 0, stream>>>(rawb, wb, proxytmp, DDIM);
    k_rownorm_to_bf16<<<CPAD, 32, 0, stream>>>(proxytmp, proxyb);
    k_class<<<NROWS, 32, 0, stream>>>(prob, conf, pcb, top1, counters);
    k_mm_abt_bf16<<<dim3(NROWS / 16, CPAD / 16), 32, 0, stream>>>(featb, proxyb, pred, CPAD);
    k_pos<<<NROWS, 32, 0, stream>>>(prob, pred, top1, conf, pos);
    k_sim<<<dim3(NROWS / 32, SLICES), 32, 0, stream>>>(featb, pcb, partM, partS);
    k_rowlse<<<NROWS / 256, 256, 0, stream>>>(partM, partS, pos, rowloss);
    k_final<<<1, 256, 0, stream>>>(rowloss, counters, out);
}
